// BroadcastingSelfAttention_73349451481722
// MI455X (gfx1250) — matrix operand order repaired and checked by reading, not compiled, not run on hardware
//
#include <hip/hip_runtime.h>

typedef __attribute__((ext_vector_type(16))) _Float16 v16h;
typedef __attribute__((ext_vector_type(8)))  float    v8f;

#define Ssz 64
#define Bsz 16
#define Fsz 64
#define Tsz 96
#define Dsz 1024
#define XROW 72  // padded LDS row stride (halfs) for Xs[f][s]

#define LOG2E 1.44269504088896340736f

// ---------------- Kernel A: score2[s,b,t] = log2(e) * softplus(sum_f X[s,b,f,t]*W[s,f] + bias[s])
// (pre-scaled by log2(e) so kernel B can use a bare v_exp_f32 per element)
__global__ __launch_bounds__(Tsz) void score_kernel(
    const float* __restrict__ X, const float* __restrict__ W,
    const float* __restrict__ bias, float* __restrict__ score2)
{
    int sb = blockIdx.x;
    int s  = sb >> 4;   // / Bsz
    int b  = sb & 15;   // % Bsz
    int t  = threadIdx.x;
    const float* xp = X + (size_t)(s*Bsz + b) * Fsz * Tsz + t;
    const float* wp = W + s * Fsz;
    float acc = bias[s];
#pragma unroll 8
    for (int f = 0; f < Fsz; ++f)
        acc = fmaf(xp[(size_t)f * Tsz], wp[f], acc);   // coalesced over t
    // numerically stable softplus
    float sp = fmaxf(acc, 0.0f) + log1pf(__expf(-fabsf(acc)));
    score2[(s*Bsz + b)*Tsz + t] = sp * LOG2E;
}

// ---------------- Kernel B: one block per (b,t); out_bt[d,f] = softmax_s(-score*dist) @ X_bt[s,f]
__global__ __launch_bounds__(256) void attn_kernel(
    const float* __restrict__ X, const float* __restrict__ dist,
    const float* __restrict__ score2, float* __restrict__ out)
{
    __shared__ _Float16 Xs[Fsz * XROW];   // X_bt transposed: [f][s], f16
    __shared__ float    Ss[Ssz];          // log2e * score_bt[s]

    const int bt  = blockIdx.x;
    const int t   = bt % Tsz;             // t fastest -> adjacent blocks write adjacent addresses
    const int b   = bt / Tsz;
    const int tid = threadIdx.x;

    // Stage X_bt (transposed, f16) and scores into LDS
    for (int i = tid; i < Ssz * Fsz; i += 256) {
        int s = i >> 6, f = i & 63;
        float x = X[((size_t)(s*Bsz + b) * Fsz + f) * Tsz + t];
        Xs[f * XROW + s] = (_Float16)x;
    }
    if (tid < Ssz)
        Ss[tid] = score2[(tid*Bsz + b)*Tsz + t];
    __syncthreads();

    const int wave = tid >> 5;
    const int lane = tid & 31;
    const int n    = lane & 15;
    const int hi   = lane >> 4;
    const int kOff = hi ? 16 : 0;   // unused: B fragments split K by kB as well
    const int kB   = hi ? 8  : 0;   // A-fragment K split across lane halves

    // B fragments (X_bt, s x f): shared by every M-tile in this block.
    // 4 N-tiles (f) x 2 K-halves (s 0..31 / 32..63), two runs of 8 halfs per lane at kB and 16+kB (16B aligned).
    v16h Bf[4][2];
#pragma unroll
    for (int nt = 0; nt < 4; ++nt)
#pragma unroll
        for (int kk = 0; kk < 2; ++kk)
            { __builtin_memcpy(&Bf[nt][kk], &Xs[(nt*16 + n) * XROW + kk*32 + kB], 16);
              __builtin_memcpy((char*)&Bf[nt][kk] + 16, &Xs[(nt*16 + n) * XROW + kk*32 + 16 + kB], 16); }

    // Each wave owns 8 M-tiles (128 d-rows); 8 waves cover D=1024.
    for (int i = 0; i < 8; ++i) {
        const int dBase = (wave * 8 + i) * 16;
        const int d     = dBase + (lane & 15);          // A row this lane supplies
        const float* dp = dist + (size_t)d * Ssz;

        // e[k] = 2^(-log2e*score[s]*dist[d,s]) for exactly the 32 K-slots this
        // lane supplies in the ISA 16-bit A layout; accumulate the row sum.
        float ev[32];
        float sum = 0.f;
#pragma unroll
        for (int g = 0; g < 4; ++g) {
            const int k0 = kB + g * 16;                 // 8 consecutive s values
            float4 p = *(const float4*)(dp + k0);
            float4 q = *(const float4*)(dp + k0 + 4);
            float e0 = __builtin_amdgcn_exp2f(-(Ss[k0+0] * p.x));
            float e1 = __builtin_amdgcn_exp2f(-(Ss[k0+1] * p.y));
            float e2 = __builtin_amdgcn_exp2f(-(Ss[k0+2] * p.z));
            float e3 = __builtin_amdgcn_exp2f(-(Ss[k0+3] * p.w));
            float e4 = __builtin_amdgcn_exp2f(-(Ss[k0+4] * q.x));
            float e5 = __builtin_amdgcn_exp2f(-(Ss[k0+5] * q.y));
            float e6 = __builtin_amdgcn_exp2f(-(Ss[k0+6] * q.z));
            float e7 = __builtin_amdgcn_exp2f(-(Ss[k0+7] * q.w));
            sum += ((e0+e1)+(e2+e3)) + ((e4+e5)+(e6+e7));
            ev[g*8+0]=e0; ev[g*8+1]=e1; ev[g*8+2]=e2; ev[g*8+3]=e3;
            ev[g*8+4]=e4; ev[g*8+5]=e5; ev[g*8+6]=e6; ev[g*8+7]=e7;
        }
        // Lanes m and m+16 each hold half the K range of row m: combine, then
        // normalize the A fragment itself (lane owns its whole row), so no
        // per-row accumulator scaling / shuffles are needed afterwards.
        float tot = sum + __shfl_xor(sum, 16);
        float inv = __builtin_amdgcn_rcpf(tot);

        v16h A0, A1;
#pragma unroll
        for (int j = 0; j < 8; ++j) {
            A0[j]     = (_Float16)(ev[j]      * inv);   // K = kB + j
            A0[8 + j] = (_Float16)(ev[8 + j]  * inv);   // K = kB + 16 + j
            A1[j]     = (_Float16)(ev[16 + j] * inv);   // K = 32 + kB + j
            A1[8 + j] = (_Float16)(ev[24 + j] * inv);   // K = 32 + kB + 16 + j
        }

        const size_t base0 = (size_t)(dBase + 8*hi) * (Bsz*Fsz*Tsz)
                           + (size_t)b * (Fsz*Tsz) + t;
#pragma unroll
        for (int nt = 0; nt < 4; ++nt) {
            v8f c = {};
            c = __builtin_amdgcn_wmma_f32_16x16x32_f16(false, A0, false, Bf[nt][0], (short)0, c, false, false);
            c = __builtin_amdgcn_wmma_f32_16x16x32_f16(false, A1, false, Bf[nt][1], (short)0, c, false, false);
            const int f = nt*16 + n;
#pragma unroll
            for (int r = 0; r < 8; ++r)
                out[base0 + (size_t)r * (Bsz*Fsz*Tsz) + (size_t)f * Tsz] = c[r];
        }
    }
}

extern "C" void kernel_launch(void* const* d_in, const int* in_sizes, int n_in,
                              void* d_out, int out_size, void* d_ws, size_t ws_size,
                              hipStream_t stream) {
    (void)in_sizes; (void)n_in; (void)out_size; (void)ws_size;
    const float* X    = (const float*)d_in[0];
    const float* dist = (const float*)d_in[1];
    const float* W    = (const float*)d_in[2];
    const float* bias = (const float*)d_in[3];
    float* out    = (float*)d_out;
    float* score2 = (float*)d_ws;   // S*B*T floats = 384 KiB scratch

    score_kernel<<<Ssz * Bsz, Tsz, 0, stream>>>(X, W, bias, score2);
    attn_kernel<<<Bsz * Tsz, 256, 0, stream>>>(X, dist, score2, out);
}
